// PCN_29308856828107
// MI455X (gfx1250) — matrix operand order repaired and checked by reading, not compiled, not run on hardware
//
#include <hip/hip_runtime.h>
#include <hip/hip_bf16.h>

// ---------------------------------------------------------------------------
// PCN inference on MI455X (gfx1250): all GEMMs via v_wmma_f32_16x16x32_bf16.
// fp32 master state, bf16 matmul operands, gamma folded into packed weights,
// update steps fused into dual-K GEMM epilogues.
// ---------------------------------------------------------------------------

typedef __bf16 bf16_t;
typedef __attribute__((ext_vector_type(16))) __bf16 v16bf;
typedef __attribute__((ext_vector_type(8)))  float  v8f;

#define LDS_STRIDE 40   // bf16 elems per LDS row (32 data + 8 pad)
#define LR_INFER 0.001f

union FragU { v16bf v; uint4 q[2]; };

// Stage a 128x32 bf16 A-tile into LDS. 256 threads: thread t -> row t/2, half t&1.
__device__ __forceinline__ void stage_A(const bf16_t* __restrict__ A, int lda,
                                        int m0, int k0, bf16_t* As) {
    int t   = threadIdx.x;
    int row = t >> 1;
    int kh  = (t & 1) * 16;
    const uint4* src = (const uint4*)(A + (size_t)(m0 + row) * lda + k0 + kh);
    uint4 a = src[0];
    uint4 b = src[1];
    uint4* dst = (uint4*)(As + row * LDS_STRIDE + kh);
    dst[0] = a;
    dst[1] = b;
}

// ---------------------------------------------------------------------------
// Main 128x128-tile GEMM. 8 waves, wave tile 32x64 (2x4 WMMA tiles).
// Supports two sequential K phases accumulating into the same accumulators
// (used for the fused update GEMMs:  acc = A1*B1 + A2*B2).
// B matrices are pre-packed into WMMA-fragment layout [kt][ntile][lane][16].
//
// MODE 0: outF = relu(acc + bias)                       (x1_pred)
// MODE 1: e = relu(acc+bias) - aux1; outF=e, outB=bf16(e)   (eps2 / eps3)
// MODE 2: xn = aux1 + LR*aux2 - LR*acc; outF=xn, outB=bf16(xn)
// MODE 3: xn = aux1 + LR*(aux2-aux1) - LR*acc  (eps1 = pred - x1_old)
// ---------------------------------------------------------------------------
template<int MODE>
__global__ __launch_bounds__(256)
void gemm128(const bf16_t* __restrict__ A1, int lda1, const bf16_t* __restrict__ B1, int K1,
             const bf16_t* __restrict__ A2, int lda2, const bf16_t* __restrict__ B2, int K2,
             int N, const float* __restrict__ bias,
             const float* __restrict__ aux1, const float* __restrict__ aux2,
             float* __restrict__ outF, bf16_t* __restrict__ outB)
{
    __shared__ __align__(16) bf16_t As[128 * LDS_STRIDE];

    const int bn     = blockIdx.x * 128;
    const int bm     = blockIdx.y * 128;
    const int w      = threadIdx.x >> 5;
    const int lane   = threadIdx.x & 31;
    const int wm     = w & 3;          // 4 wave rows   (32 M each)
    const int wn     = w >> 2;         // 2 wave cols   (64 N each)
    const int laneLo = lane & 15;
    const int laneHi = lane >> 4;
    const int NT     = N >> 4;         // total 16-wide N tiles in packed B

    const v8f vzero = {0.f, 0.f, 0.f, 0.f, 0.f, 0.f, 0.f, 0.f};
    v8f acc[2][4];
#pragma unroll
    for (int mt = 0; mt < 2; ++mt)
#pragma unroll
        for (int nt = 0; nt < 4; ++nt) acc[mt][nt] = vzero;

    for (int phase = 0; phase < 2; ++phase) {
        const bf16_t* A  = phase ? A2 : A1;
        const bf16_t* Bp = phase ? B2 : B1;
        const int lda    = phase ? lda2 : lda1;
        const int K      = phase ? K2 : K1;
        if (K == 0) continue;
        const int KT = K >> 5;

        for (int kt = 0; kt < KT; ++kt) {
            __syncthreads();
            stage_A(A, lda, bm, kt * 32, As);
            __syncthreads();

            // A fragments (two 16x32 sub-tiles) from LDS: 2x ds_load_b128 each
            FragU af[2];
#pragma unroll
            for (int mt = 0; mt < 2; ++mt) {
                int r = wm * 32 + mt * 16 + laneLo;
                af[mt].q[0] = *(const uint4*)(As + r * LDS_STRIDE + laneHi * 8);
                af[mt].q[1] = *(const uint4*)(As + r * LDS_STRIDE + 16 + laneHi * 8);
            }
            // B fragments straight from pre-packed global (L2-resident)
#pragma unroll
            for (int nt = 0; nt < 4; ++nt) {
                int ntile = (bn + wn * 64 + nt * 16) >> 4;
                const uint4* bp =
                    (const uint4*)(Bp + (((size_t)kt * NT + ntile) * 32 + lane) * 16);
                FragU bfrag;
                bfrag.q[0] = bp[0];
                bfrag.q[1] = bp[1];
#pragma unroll
                for (int mt = 0; mt < 2; ++mt)
                    acc[mt][nt] = __builtin_amdgcn_wmma_f32_16x16x32_bf16(
                        false, af[mt].v, false, bfrag.v, (short)0, acc[mt][nt],
                        false, false);
            }
        }
    }

    // Epilogue. C layout: VGPR v -> row = v + 8*laneHi, col = lane&15.
#pragma unroll
    for (int mt = 0; mt < 2; ++mt) {
#pragma unroll
        for (int nt = 0; nt < 4; ++nt) {
#pragma unroll
            for (int v = 0; v < 8; ++v) {
                int row = bm + wm * 32 + mt * 16 + laneHi * 8 + v;
                int col = bn + wn * 64 + nt * 16 + laneLo;
                size_t idx = (size_t)row * N + col;
                float a = acc[mt][nt][v];
                if (MODE == 0) {
                    float val = a + bias[col];
                    outF[idx] = val > 0.f ? val : 0.f;
                } else if (MODE == 1) {
                    float val = a + bias[col];
                    val = val > 0.f ? val : 0.f;
                    float e = val - aux1[idx];
                    outF[idx] = e;
                    outB[idx] = (bf16_t)e;
                } else {
                    float xo  = aux1[idx];
                    float eps = (MODE == 3) ? (aux2[idx] - xo) : aux2[idx];
                    float xn  = xo + LR_INFER * eps - LR_INFER * a;
                    outF[idx] = xn;
                    outB[idx] = (bf16_t)xn;
                }
            }
        }
    }
}

// ---------------------------------------------------------------------------
// Narrow-N GEMM (N = 16 tile). 8 waves x 16 rows = 128-row blocks.
// SMODE 0: x4_pred -> bf16 [M][32], K-padded with zeros (cols 16..31 = 0)
// SMODE 1: final output: out[row*10+col] = acc + b4[col]  (col < 10)
// ---------------------------------------------------------------------------
template<int SMODE>
__global__ __launch_bounds__(256)
void gemm_n16(const bf16_t* __restrict__ A, int lda, const bf16_t* __restrict__ Bp, int K,
              const float* __restrict__ bias,
              float* __restrict__ outF, bf16_t* __restrict__ outB)
{
    __shared__ __align__(16) bf16_t As[128 * LDS_STRIDE];

    const int bm     = blockIdx.x * 128;
    const int w      = threadIdx.x >> 5;
    const int lane   = threadIdx.x & 31;
    const int laneLo = lane & 15;
    const int laneHi = lane >> 4;

    const v8f vzero = {0.f, 0.f, 0.f, 0.f, 0.f, 0.f, 0.f, 0.f};
    v8f acc = vzero;

    const int KT = K >> 5;
    for (int kt = 0; kt < KT; ++kt) {
        __syncthreads();
        stage_A(A, lda, bm, kt * 32, As);
        __syncthreads();

        int r = w * 16 + laneLo;
        FragU af;
        af.q[0] = *(const uint4*)(As + r * LDS_STRIDE + laneHi * 8);
        af.q[1] = *(const uint4*)(As + r * LDS_STRIDE + 16 + laneHi * 8);

        const uint4* bp = (const uint4*)(Bp + ((size_t)kt * 32 + lane) * 16); // ntile 0
        FragU bfrag;
        bfrag.q[0] = bp[0];
        bfrag.q[1] = bp[1];

        acc = __builtin_amdgcn_wmma_f32_16x16x32_bf16(
            false, af.v, false, bfrag.v, (short)0, acc, false, false);
    }

#pragma unroll
    for (int v = 0; v < 8; ++v) {
        int row = bm + w * 16 + laneHi * 8 + v;
        int col = laneLo;
        float bv  = (col < 10) ? bias[col] : 0.f;
        float val = acc[v] + bv;          // cols 10..15: packed B was zero => 0
        if (SMODE == 0) {
            outB[(size_t)row * 32 + col]      = (bf16_t)val;
            outB[(size_t)row * 32 + col + 16] = (bf16_t)0.f;  // K-pad to 32
        } else {
            if (col < 10) outF[(size_t)row * 10 + col] = val;
        }
    }
}

// ---------------------------------------------------------------------------
// Pack an fp32 weight into bf16 WMMA B-fragment layout [kt][nt][lane][16],
// with optional transpose, scale (used to fold -gamma), and zero padding.
// Logical B[k][n] = scale * (doT ? src[n][k] : src[k][n]).
// Per-lane element e: k = kt*32 + (lane>>4)*8 + (e<8 ? e : e+8), n = nt*16 + (lane&15).
// ---------------------------------------------------------------------------
__global__ void pack_weight(const float* __restrict__ src, int srcR, int srcC,
                            int doT, float scale,
                            bf16_t* __restrict__ dst, int K, int N)
{
    int tid = blockIdx.x * blockDim.x + threadIdx.x;
    int NT = N >> 4;
    int total = (K >> 5) * NT * 32;
    if (tid >= total) return;
    int lane = tid & 31;
    int g    = tid >> 5;
    int nt   = g % NT;
    int kt   = g / NT;
    int n     = nt * 16 + (lane & 15);
    int kbase = kt * 32 + (lane >> 4) * 8;
    bf16_t* d = dst + (size_t)tid * 16;
#pragma unroll
    for (int e = 0; e < 16; ++e) {
        int k  = kbase + (e < 8 ? e : e + 8);
        int sr = doT ? n : k;
        int sc = doT ? k : n;
        float v = 0.f;
        if (sr < srcR && sc < srcC) v = src[(size_t)sr * srcC + sc] * scale;
        d[e] = (bf16_t)v;
    }
}

// fp32 -> bf16 with column padding (x: 784 -> 800)
__global__ void convert_pad(const float* __restrict__ src, int cols,
                            bf16_t* __restrict__ dst, int dcols, size_t total)
{
    size_t tid = (size_t)blockIdx.x * blockDim.x + threadIdx.x;
    if (tid >= total) return;
    size_t r = tid / dcols;
    int    c = (int)(tid % dcols);
    float v = (c < cols) ? src[r * cols + c] : 0.f;
    dst[tid] = (bf16_t)v;
}

__global__ void zero_u4(uint4* __restrict__ p, size_t n16)
{
    size_t i = (size_t)blockIdx.x * blockDim.x + threadIdx.x;
    if (i < n16) { uint4 z = {0u, 0u, 0u, 0u}; p[i] = z; }
}

// ---------------------------------------------------------------------------
// Host-side orchestration
// ---------------------------------------------------------------------------
extern "C" void kernel_launch(void* const* d_in, const int* in_sizes, int n_in,
                              void* d_out, int out_size, void* d_ws, size_t ws_size,
                              hipStream_t stream)
{
    (void)in_sizes; (void)n_in; (void)out_size; (void)ws_size;
    constexpr int Bn = 8192, DIN = 784, DINP = 800;
    constexpr int H1 = 512, H2 = 1024, H3 = 512;
    constexpr int NSTEPS = 50;

    const float* x  = (const float*)d_in[0];
    const float* W1 = (const float*)d_in[1];
    const float* W2 = (const float*)d_in[2];
    const float* W3 = (const float*)d_in[3];
    const float* W4 = (const float*)d_in[4];
    const float* b1 = (const float*)d_in[5];
    const float* b2 = (const float*)d_in[6];
    const float* b3 = (const float*)d_in[7];
    const float* b4 = (const float*)d_in[8];
    const float* L1 = (const float*)d_in[9];
    const float* L2 = (const float*)d_in[10];
    const float* L3 = (const float*)d_in[11];

    char* ws = (char*)d_ws;
    size_t off = 0;
    auto alloc = [&](size_t bytes) -> void* {
        void* p = ws + off;
        off = (off + bytes + 255) & ~(size_t)255;
        return p;
    };

    bf16_t* xb   = (bf16_t*)alloc((size_t)Bn * DINP * 2);
    float*  x1p  = (float*) alloc((size_t)Bn * H1 * 4);
    float*  x1f[2] = { (float*)alloc((size_t)Bn * H1 * 4), (float*)alloc((size_t)Bn * H1 * 4) };
    float*  x2f[2] = { (float*)alloc((size_t)Bn * H2 * 4), (float*)alloc((size_t)Bn * H2 * 4) };
    float*  x3f[2] = { (float*)alloc((size_t)Bn * H3 * 4), (float*)alloc((size_t)Bn * H3 * 4) };
    bf16_t* x1b[2] = { (bf16_t*)alloc((size_t)Bn * H1 * 2), (bf16_t*)alloc((size_t)Bn * H1 * 2) };
    bf16_t* x2b[2] = { (bf16_t*)alloc((size_t)Bn * H2 * 2), (bf16_t*)alloc((size_t)Bn * H2 * 2) };
    bf16_t* x3b[2] = { (bf16_t*)alloc((size_t)Bn * H3 * 2), (bf16_t*)alloc((size_t)Bn * H3 * 2) };
    float*  eps2f = (float*) alloc((size_t)Bn * H2 * 4);
    float*  eps3f = (float*) alloc((size_t)Bn * H3 * 4);
    bf16_t* eps2b = (bf16_t*)alloc((size_t)Bn * H2 * 2);
    bf16_t* eps3b = (bf16_t*)alloc((size_t)Bn * H3 * 2);
    bf16_t* x4pb  = (bf16_t*)alloc((size_t)Bn * 32 * 2);

    bf16_t* W1p  = (bf16_t*)alloc((size_t)DINP * H1 * 2);
    bf16_t* W2p  = (bf16_t*)alloc((size_t)H1 * H2 * 2);
    bf16_t* W3p  = (bf16_t*)alloc((size_t)H2 * H3 * 2);
    bf16_t* W4p  = (bf16_t*)alloc((size_t)H3 * 16 * 2);
    bf16_t* W4Tp = (bf16_t*)alloc((size_t)32 * H3 * 2);
    bf16_t* W3Tp = (bf16_t*)alloc((size_t)H3 * H2 * 2);
    bf16_t* W2Tp = (bf16_t*)alloc((size_t)H2 * H1 * 2);
    bf16_t* L1s  = (bf16_t*)alloc((size_t)H1 * H1 * 2);
    bf16_t* L2s  = (bf16_t*)alloc((size_t)H2 * H2 * 2);
    bf16_t* L3s  = (bf16_t*)alloc((size_t)H3 * H3 * 2);

    auto zero = [&](void* p, size_t bytes) {
        size_t n16 = bytes / 16;
        zero_u4<<<dim3((unsigned)((n16 + 255) / 256)), 256, 0, stream>>>((uint4*)p, n16);
    };
    // initial state x1 = x2 = x3 = 0 (fp32 and bf16, parity-0 buffers)
    zero(x1f[0], (size_t)Bn * H1 * 4);
    zero(x2f[0], (size_t)Bn * H2 * 4);
    zero(x3f[0], (size_t)Bn * H3 * 4);
    zero(x1b[0], (size_t)Bn * H1 * 2);
    zero(x2b[0], (size_t)Bn * H2 * 2);
    zero(x3b[0], (size_t)Bn * H3 * 2);

    { // x -> bf16, K padded 784 -> 800
        size_t total = (size_t)Bn * DINP;
        convert_pad<<<dim3((unsigned)((total + 255) / 256)), 256, 0, stream>>>(
            x, DIN, xb, DINP, total);
    }

    auto pack = [&](const float* src, int sr, int sc, int doT, float scale,
                    bf16_t* dst, int K, int N) {
        int total = (K >> 5) * (N >> 4) * 32;
        pack_weight<<<dim3((unsigned)((total + 255) / 256)), 256, 0, stream>>>(
            src, sr, sc, doT, scale, dst, K, N);
    };
    pack(W1, DIN, H1, 0, 1.0f,  W1p,  DINP, H1);
    pack(W2, H1,  H2, 0, 1.0f,  W2p,  H1,   H2);
    pack(W3, H2,  H3, 0, 1.0f,  W3p,  H2,   H3);
    pack(W4, H3,  10, 0, 1.0f,  W4p,  H3,   16);
    pack(W4, H3,  10, 1, 1.0f,  W4Tp, 32,   H3);
    pack(W3, H2,  H3, 1, 1.0f,  W3Tp, H3,   H2);
    pack(W2, H1,  H2, 1, 1.0f,  W2Tp, H2,   H1);
    pack(L1, H1,  H1, 0, -0.1f, L1s,  H1,   H1);
    pack(L2, H2,  H2, 0, -0.1f, L2s,  H2,   H2);
    pack(L3, H3,  H3, 0, -0.1f, L3s,  H3,   H3);

    const dim3 blk(256);
    const unsigned MB = Bn / 128;

    // x1_pred = relu(x @ W1 + b1)
    gemm128<0><<<dim3(H1 / 128, MB), blk, 0, stream>>>(
        xb, DINP, W1p, DINP, nullptr, 0, nullptr, 0,
        H1, b1, nullptr, nullptr, x1p, nullptr);

    for (int s = 0; s < NSTEPS; ++s) {
        int R = s & 1, Wd = 1 - R;
        // eps2 = relu(x1 @ W2 + b2) - x2
        gemm128<1><<<dim3(H2 / 128, MB), blk, 0, stream>>>(
            x1b[R], H1, W2p, H1, nullptr, 0, nullptr, 0,
            H2, b2, x2f[R], nullptr, eps2f, eps2b);
        // eps3 = relu(x2 @ W3 + b3) - x3
        gemm128<1><<<dim3(H3 / 128, MB), blk, 0, stream>>>(
            x2b[R], H2, W3p, H2, nullptr, 0, nullptr, 0,
            H3, b3, x3f[R], nullptr, eps3f, eps3b);
        // x4_pred = x3 @ W4 + b4  (bf16, K-padded to 32)
        gemm_n16<0><<<dim3(MB), blk, 0, stream>>>(
            x3b[R], H3, W4p, H3, b4, nullptr, x4pb);
        // x3 <- x3 + LR*eps3 - LR*(x4_pred @ W4T + x3 @ (-g*L3))
        gemm128<2><<<dim3(H3 / 128, MB), blk, 0, stream>>>(
            x4pb, 32, W4Tp, 32, x3b[R], H3, L3s, H3,
            H3, nullptr, x3f[R], eps3f, x3f[Wd], x3b[Wd]);
        // x2 <- x2 + LR*eps2 - LR*(eps3 @ W3T + x2 @ (-g*L2))
        gemm128<2><<<dim3(H2 / 128, MB), blk, 0, stream>>>(
            eps3b, H3, W3Tp, H3, x2b[R], H2, L2s, H2,
            H2, nullptr, x2f[R], eps2f, x2f[Wd], x2b[Wd]);
        // x1 <- x1 + LR*(x1_pred - x1) - LR*(eps2 @ W2T + x1 @ (-g*L1))
        gemm128<3><<<dim3(H1 / 128, MB), blk, 0, stream>>>(
            eps2b, H2, W2Tp, H2, x1b[R], H1, L1s, H1,
            H1, nullptr, x1f[R], x1p, x1f[Wd], x1b[Wd]);
    }

    // out = x3_final @ W4 + b4   (50 steps even -> final state in parity 0)
    gemm_n16<1><<<dim3(MB), blk, 0, stream>>>(
        x3b[0], H3, W4p, H3, b4, (float*)d_out, nullptr);
}
